// LSTMModel_18064632447273
// MI455X (gfx1250) — matrix operand order repaired and checked by reading, not compiled, not run on hardware
//
#include <hip/hip_runtime.h>

typedef __attribute__((ext_vector_type(16))) _Float16 v16h;
typedef __attribute__((ext_vector_type(8)))  float    v8f;

#define B_  1024
#define T_  1024
#define H_  20
#define G_  80     // 4*H
#define NT_ 5      // G/16 column tiles
#define OUT_ 2

__device__ __forceinline__ float sigmoidf_(float x) {
    return 1.0f / (1.0f + __expf(-x));
}

// One wave (32 threads) owns one 16-row batch tile for the whole sequence.
// Per timestep, per 16-gate column tile: two chained WMMAs
//   acc = bias; acc = x_t * W_ih^T + acc; acc = h * W_hh^T + acc
// K = IN_DIM (resp. H) zero-padded to 32 inside the f16 fragments.
// x_t tiles are double-buffered in LDS so each step needs only two
// wave-local sync points (gates->elementwise, h/x->next A build).
template<int IN_DIM, bool IN_IS_F16>
__global__ __launch_bounds__(32)
void lstm_layer(const void* __restrict__ xin,
                const float* __restrict__ w_ih,
                const float* __restrict__ w_hh,
                const float* __restrict__ b_ih,
                const float* __restrict__ b_hh,
                _Float16* __restrict__ out_seq)
{
    __shared__ _Float16 xh[2][16][32];  // double-buffered x_t tile, K-padded
    __shared__ _Float16 hb[16][32];     // current h tile, K-padded
    __shared__ float    gates[16][G_];
    __shared__ float    cb[16][H_];

    const int lane = threadIdx.x;       // 0..31
    const int b0   = blockIdx.x * 16;   // batch tile base (64 tiles)

    const float*    xin_f32 = (const float*)xin;
    const _Float16* xin_f16 = (const _Float16*)xin;

    // ---- init LDS (padding columns stay 0 forever) ----
    for (int idx = lane; idx < 16 * 32; idx += 32) {
        const int r = idx >> 5, k = idx & 31;
        xh[0][r][k] = (_Float16)0.0f;
        xh[1][r][k] = (_Float16)0.0f;
        hb[r][k]    = (_Float16)0.0f;
    }
    for (int idx = lane; idx < 16 * H_; idx += 32)
        cb[idx / H_][idx % H_] = 0.0f;

    // ---- persistent weight B-fragments + bias C-fragments (live in VGPRs) ----
    // B 32x16 f16 layout: lane<16 -> N=lane,    j=0..7 -> K=j,   j=8..15 -> K=16+(j-8)
    //                     lane>=16-> N=lane-16, j=0..7 -> K=8+j, j=8..15 -> K=24+(j-8)
    const int ncol  = lane & 15;
    const int kofsB = (lane >= 16) ? 8 : 0;

    v16h Bih[NT_], Bhh[NT_];
    v8f  biasC[NT_];
#pragma unroll
    for (int nt = 0; nt < NT_; ++nt) {
        const int n = nt * 16 + ncol;            // gate index 0..79
        v16h bi, bh;
#pragma unroll
        for (int j = 0; j < 16; ++j) {
            const int k = kofsB + j + ((j >= 8) ? 8 : 0);
            bi[j] = (k < IN_DIM) ? (_Float16)w_ih[n * IN_DIM + k] : (_Float16)0.0f;
            bh[j] = (k < H_)     ? (_Float16)w_hh[n * H_    + k] : (_Float16)0.0f;
        }
        Bih[nt] = bi;
        Bhh[nt] = bh;
        const float bs = b_ih[n] + b_hh[n];
        v8f bc;
#pragma unroll
        for (int r = 0; r < 8; ++r) bc[r] = bs;
        biasC[nt] = bc;
    }

    // ---- load x_0 into buffer 0 ----
    for (int idx = lane; idx < 16 * IN_DIM; idx += 32) {
        const int r = idx / IN_DIM, k = idx % IN_DIM;
        const size_t src = ((size_t)(b0 + r) * T_ + 0) * IN_DIM + k;
        const float v = IN_IS_F16 ? (float)xin_f16[src] : xin_f32[src];
        xh[0][r][k] = (_Float16)v;
    }
    __syncthreads();

    // A 16x32 f16 layout: row M = lane&15;
    //   lane<16 : j=0..7 -> K=j,   j=8..15 -> K=16+(j-8)
    //   lane>=16: j=0..7 -> K=8+j, j=8..15 -> K=24+(j-8)
    const int kofsA = (lane >= 16) ? 8 : 0;
    const int mrow  = lane & 15;
    const int mbase = (lane >> 4) * 8;           // C/D fragment row base

    for (int t = 0; t < T_; ++t) {
        const int cur = t & 1, nxt = cur ^ 1;

        // 1) build A fragments from LDS
        v16h Ax, Ah;
#pragma unroll
        for (int j = 0; j < 8; ++j) {
            const int k0 = kofsA + j;
            const int k1 = kofsA + 16 + j;
            Ax[j]     = xh[cur][mrow][k0];
            Ax[j + 8] = xh[cur][mrow][k1];
            Ah[j]     = hb[mrow][k0];
            Ah[j + 8] = hb[mrow][k1];
        }

        // 2) gates = bias + x_t*W_ih^T + h*W_hh^T   (10 WMMAs per step)
#pragma unroll
        for (int nt = 0; nt < NT_; ++nt) {
            v8f acc = biasC[nt];
            acc = __builtin_amdgcn_wmma_f32_16x16x32_f16(
                      false, Ax, false, Bih[nt], (short)0, acc, false, false);
            acc = __builtin_amdgcn_wmma_f32_16x16x32_f16(
                      false, Ah, false, Bhh[nt], (short)0, acc, false, false);
            const int n = nt * 16 + ncol;
#pragma unroll
            for (int r = 0; r < 8; ++r)
                gates[mbase + r][n] = acc[r];
        }
        __syncthreads();   // gates visible; xh[cur]/hb reads done

        // 3) overlap: load x_{t+1} into the other buffer (disjoint LDS)
        if (t + 1 < T_) {
            for (int idx = lane; idx < 16 * IN_DIM; idx += 32) {
                const int r = idx / IN_DIM, k = idx % IN_DIM;
                const size_t src = ((size_t)(b0 + r) * T_ + (t + 1)) * IN_DIM + k;
                const float v = IN_IS_F16 ? (float)xin_f16[src] : xin_f32[src];
                xh[nxt][r][k] = (_Float16)v;
            }
            if (t + 2 < T_) {   // prefetch the step after that into near caches
                const size_t pf = ((size_t)(b0 + mrow) * T_ + (t + 2)) * IN_DIM;
                const char* p = IN_IS_F16 ? (const char*)(xin_f16 + pf)
                                          : (const char*)(xin_f32 + pf);
                __builtin_prefetch(p, 0, 3);
            }
        }

        // 4) elementwise cell update: 16*20 units / 32 lanes = 10 per lane
        for (int idx = lane; idx < 16 * H_; idx += 32) {
            const int m = idx / H_, h = idx % H_;
            const float gi = gates[m][h];
            const float gf = gates[m][H_ + h];
            const float gg = gates[m][2 * H_ + h];
            const float go = gates[m][3 * H_ + h];
            const float c  = sigmoidf_(gf) * cb[m][h] + sigmoidf_(gi) * tanhf(gg);
            cb[m][h] = c;
            const float hv = sigmoidf_(go) * tanhf(c);
            hb[m][h] = (_Float16)hv;
            out_seq[((size_t)(b0 + m) * T_ + t) * H_ + h] = (_Float16)hv;
        }
        __syncthreads();   // hb / xh[nxt] visible for next A build
    }
}

// Tiny epilogue: out[b,o] = h_last[b,:] . fc_w[o,:] + fc_b[o]
__global__ __launch_bounds__(256)
void fc_kernel(const _Float16* __restrict__ seq,
               const float* __restrict__ fc_w,
               const float* __restrict__ fc_b,
               float* __restrict__ out)
{
    const int b = blockIdx.x * blockDim.x + threadIdx.x;
    if (b >= B_) return;
    const _Float16* hlast = &seq[((size_t)b * T_ + (T_ - 1)) * H_];
    float h[H_];
#pragma unroll
    for (int i = 0; i < H_; ++i) h[i] = (float)hlast[i];
#pragma unroll
    for (int o = 0; o < OUT_; ++o) {
        float acc = fc_b[o];
#pragma unroll
        for (int i = 0; i < H_; ++i) acc += h[i] * fc_w[o * H_ + i];
        out[b * OUT_ + o] = acc;
    }
}

extern "C" void kernel_launch(void* const* d_in, const int* in_sizes, int n_in,
                              void* d_out, int out_size, void* d_ws, size_t ws_size,
                              hipStream_t stream)
{
    const float* x     = (const float*)d_in[0];
    const float* w_ih0 = (const float*)d_in[1];
    const float* w_hh0 = (const float*)d_in[2];
    const float* b_ih0 = (const float*)d_in[3];
    const float* b_hh0 = (const float*)d_in[4];
    const float* w_ih1 = (const float*)d_in[5];
    const float* w_hh1 = (const float*)d_in[6];
    const float* b_ih1 = (const float*)d_in[7];
    const float* b_hh1 = (const float*)d_in[8];
    const float* w_ih2 = (const float*)d_in[9];
    const float* w_hh2 = (const float*)d_in[10];
    const float* b_ih2 = (const float*)d_in[11];
    const float* b_hh2 = (const float*)d_in[12];
    const float* fc_w  = (const float*)d_in[13];
    const float* fc_b  = (const float*)d_in[14];

    // ping-pong inter-layer hidden sequences, f16 [B,T,H] (2 x 40 MiB, L2-resident)
    _Float16* seqA = (_Float16*)d_ws;
    _Float16* seqB = seqA + (size_t)B_ * T_ * H_;

    const dim3 grid(B_ / 16), block(32);
    lstm_layer<1,  false><<<grid, block, 0, stream>>>(x,    w_ih0, w_hh0, b_ih0, b_hh0, seqA);
    lstm_layer<H_, true ><<<grid, block, 0, stream>>>(seqA, w_ih1, w_hh1, b_ih1, b_hh1, seqB);
    lstm_layer<H_, true ><<<grid, block, 0, stream>>>(seqB, w_ih2, w_hh2, b_ih2, b_hh2, seqA);
    fc_kernel<<<dim3((B_ + 255) / 256), dim3(256), 0, stream>>>(seqA, fc_w, fc_b, (float*)d_out);
}
